// DopDense_89489938579795
// MI455X (gfx1250) — matrix operand order repaired and checked by reading, not compiled, not run on hardware
//
#include <hip/hip_runtime.h>
#include <hip/hip_bf16.h>

typedef float    v2f  __attribute__((ext_vector_type(2)));
typedef float    v8f  __attribute__((ext_vector_type(8)));
typedef _Float16 v4h  __attribute__((ext_vector_type(4)));
typedef _Float16 v8h  __attribute__((ext_vector_type(8)));
typedef _Float16 v16h __attribute__((ext_vector_type(16)));

#define UNITS 512
#define N_IN  512
#define N_DOP 128
#define BATCH 65536
#define MROWS 32              // M rows per workgroup (2 M-tiles per wave)

// ---------------------------------------------------------------------------
// Kernel 1: per-column multiplier vector mult[512] (exact reference semantics,
// incl. np.linspace(1,511,128,int32) truncation and the column-0 case that
// receives both left-fac(d=1) and right-fac(d=511) in reference order).
// ---------------------------------------------------------------------------
__global__ __launch_bounds__(N_DOP) void dop_mult_kernel(
    const float* __restrict__ w, const float* __restrict__ old_w,
    const float* __restrict__ indicator, const int* __restrict__ batch_ctr,
    float* __restrict__ mult)
{
    __shared__ int   dop[N_DOP];
    __shared__ float lf[N_DOP];
    __shared__ float rf[N_DOP];

    const int t = threadIdx.x;
    const double step = 510.0 / 127.0;              // (511-1)/(128-1)
    int d = (t == N_DOP - 1) ? (UNITS - 1) : (int)(1.0 + step * (double)t);
    dop[t] = d;
    __syncthreads();

    float s = 0.0f;
    for (int i = 0; i < N_IN; ++i)
        s += fabsf(w[i * UNITS + d] - old_w[i * UNITS + d]);
    const float diff = s * (1.0f / (float)N_IN);

    const bool active = (diff > 0.0f) &&
                        (((float)(*batch_ctr) - indicator[t]) > 2.0f);
    const float fac = 1.0f + 10.0f * diff;

    bool lok = true, rok = true;
    for (int j = 0; j < N_DOP; ++j) {
        const int dj = dop[j];
        lok = lok && (dj != d - 1);
        rok = rok && (dj != d + 1);
    }
    lf[t] = (active && lok) ? fac : 1.0f;
    rf[t] = (active && rok) ? fac : 1.0f;
    __syncthreads();

    for (int c = t * 4; c < t * 4 + 4; ++c) {
        float m = 1.0f;
        for (int j = 0; j < N_DOP; ++j) {
            const int dj = dop[j];
            if (((dj - 1) & (UNITS - 1)) == c) m *= lf[j];
            if (((dj + 1) & (UNITS - 1)) == c) m *= rf[j];
        }
        mult[c] = m;
    }
}

// ---------------------------------------------------------------------------
// Kernel 2: split w_new = w * mult into f16 hi/lo, stored fragment-packed for
// the 32x16 f16 B layout (ISA 7.12.2): for K-block kb = k/32, column n,
// lane-half h = (k>>3)&1, the 16 f16 values K = kb*32 + h*8 + {0..7}, kb*32 + 16 + h*8 + {0..7}
// are contiguous -> each B fragment is one aligned 32B run per lane.
//   p = kb*(512*32) + n*32 + h*16 + ((k>>4)&1)*8 + (k&7)
// ---------------------------------------------------------------------------
__global__ __launch_bounds__(256) void pack_w_kernel(
    const float* __restrict__ w, const float* __restrict__ mult,
    _Float16* __restrict__ whi, _Float16* __restrict__ wlo)
{
    const int i = blockIdx.x * 256 + threadIdx.x;   // 0 .. 512*512-1
    const int k = i >> 9;
    const int n = i & (UNITS - 1);
    const float val = w[i] * mult[n];
    const _Float16 h = (_Float16)val;
    const _Float16 l = (_Float16)(val - (float)h);
    const int p = (k >> 5) * (UNITS * 32) + n * 32 + ((k >> 3) & 1) * 16 + ((k >> 4) & 1) * 8 + (k & 7);
    whi[p] = h;
    wlo[p] = l;
}

// ---------------------------------------------------------------------------
// Kernel 3: out = relu(x @ w_new + b) via 3x f16 WMMA (Markidis split):
//   x@w ~= xhi@whi + xlo@whi + xhi@wlo   (fp32 accumulate, ~2^-22 rel. err)
//
// 256 threads = 8 waves per block; block owns 32 rows x all 512 cols.
// x slab read once from HBM as fp32 and split in-register into two f16 LDS
// slabs (32KB + 32KB). Wave i owns 64 cols: 2 M-tiles x 4 N-tiles fp32
// accumulators. Per K=32 step: 8 ds_load_b128 (A) + 8 v16h B loads feed
// 24 v_wmma_f32_16x16x32_f16.
//
// A 16x32 f16 fragment (lane: row=L&15, half=L>>4): halves at
//   k0 + half*8 + {0..7}  and  k0 + 16 + half*8 + {0..7}
// ---------------------------------------------------------------------------
static __device__ inline v16h load_a_frag(const _Float16* base, int k0)
{
    const v8h c1 = *(const v8h*)(base + k0);
    const v8h c2 = *(const v8h*)(base + k0 + 16);
    return __builtin_shufflevector(c1, c2, 0, 1, 2, 3, 4, 5, 6, 7,
                                           8, 9, 10, 11, 12, 13, 14, 15);
}

__global__ __launch_bounds__(256) void dop_gemm_kernel(
    const float* __restrict__ x,
    const _Float16* __restrict__ whi, const _Float16* __restrict__ wlo,
    const float* __restrict__ bias, float* __restrict__ out)
{
    __shared__ _Float16 xhi[MROWS * N_IN];   // 32 KB
    __shared__ _Float16 xlo[MROWS * N_IN];   // 32 KB

    const int tid  = threadIdx.x;
    const int wave = tid >> 5;
    const int lane = tid & 31;
    const int m0   = blockIdx.x * MROWS;

    // stage + split: 4096 float4 per block, 16 per thread
    {
        const float4* xg = (const float4*)(x + (size_t)m0 * N_IN);
        v4h* xh4 = (v4h*)xhi;
        v4h* xl4 = (v4h*)xlo;
        #pragma unroll
        for (int i = 0; i < (MROWS * N_IN / 4) / 256; ++i) {
            const int idx = tid + i * 256;
            const float4 v = xg[idx];
            v4h h, l;
            h.x = (_Float16)v.x;  l.x = (_Float16)(v.x - (float)h.x);
            h.y = (_Float16)v.y;  l.y = (_Float16)(v.y - (float)h.y);
            h.z = (_Float16)v.z;  l.z = (_Float16)(v.z - (float)h.z);
            h.w = (_Float16)v.w;  l.w = (_Float16)(v.w - (float)h.w);
            xh4[idx] = h;
            xl4[idx] = l;
        }
    }
    __syncthreads();

    const int row  = lane & 15;
    const int ncol = lane & 15;
    const int half = lane >> 4;
    const int n0   = wave * 64;

    v8f acc[2][4] = {};

    const _Float16* ah0 = xhi + row * N_IN + half * 8;          // M-tile 0
    const _Float16* al0 = xlo + row * N_IN + half * 8;
    const _Float16* ah1 = xhi + (16 + row) * N_IN + half * 8;   // M-tile 1
    const _Float16* al1 = xlo + (16 + row) * N_IN + half * 8;

    // B fragment base: column (n0 + nt*16 + ncol), this lane-half
    const size_t boff = (size_t)(n0 + ncol) * 32 + half * 16;

    for (int k0 = 0; k0 < N_IN; k0 += 32) {
        const v16h fah0 = load_a_frag(ah0, k0);
        const v16h fal0 = load_a_frag(al0, k0);
        const v16h fah1 = load_a_frag(ah1, k0);
        const v16h fal1 = load_a_frag(al1, k0);

        const size_t kboff = (size_t)(k0 >> 5) * (UNITS * 32) + boff;
        #pragma unroll
        for (int nt = 0; nt < 4; ++nt) {
            const v16h bh = *(const v16h*)(whi + kboff + nt * (16 * 32));
            const v16h bl = *(const v16h*)(wlo + kboff + nt * (16 * 32));

            acc[0][nt] = __builtin_amdgcn_wmma_f32_16x16x32_f16(
                false, fah0, false, bh, (short)0, acc[0][nt], false, false);
            acc[0][nt] = __builtin_amdgcn_wmma_f32_16x16x32_f16(
                false, fal0, false, bh, (short)0, acc[0][nt], false, false);
            acc[0][nt] = __builtin_amdgcn_wmma_f32_16x16x32_f16(
                false, fah0, false, bl, (short)0, acc[0][nt], false, false);

            acc[1][nt] = __builtin_amdgcn_wmma_f32_16x16x32_f16(
                false, fah1, false, bh, (short)0, acc[1][nt], false, false);
            acc[1][nt] = __builtin_amdgcn_wmma_f32_16x16x32_f16(
                false, fal1, false, bh, (short)0, acc[1][nt], false, false);
            acc[1][nt] = __builtin_amdgcn_wmma_f32_16x16x32_f16(
                false, fah1, false, bl, (short)0, acc[1][nt], false, false);
        }
    }

    // epilogue: bias + ReLU (mult already folded into packed w)
    const int mbase = half * 8;
    #pragma unroll
    for (int nt = 0; nt < 4; ++nt) {
        const int   n  = n0 + nt * 16 + ncol;
        const float bv = bias[n];
        #pragma unroll
        for (int v = 0; v < 8; ++v) {
            const float v0 = acc[0][nt][v] + bv;
            const float v1 = acc[1][nt][v] + bv;
            out[(size_t)(m0 + mbase + v) * UNITS + n]      = fmaxf(v0, 0.0f);
            out[(size_t)(m0 + 16 + mbase + v) * UNITS + n] = fmaxf(v1, 0.0f);
        }
    }
}

// ---------------------------------------------------------------------------
extern "C" void kernel_launch(void* const* d_in, const int* in_sizes, int n_in,
                              void* d_out, int out_size, void* d_ws, size_t ws_size,
                              hipStream_t stream) {
    const float* x     = (const float*)d_in[0];
    const float* w     = (const float*)d_in[1];
    const float* b     = (const float*)d_in[2];
    const float* old_w = (const float*)d_in[3];
    const float* ind   = (const float*)d_in[4];
    const int*   bc    = (const int*)d_in[5];

    // workspace layout: mult[512] f32 | whi[512*512] f16 | wlo[512*512] f16
    float*    mult = (float*)d_ws;
    _Float16* whi  = (_Float16*)((char*)d_ws + 2048);
    _Float16* wlo  = whi + (size_t)UNITS * N_IN;
    float*    out  = (float*)d_out;

    dop_mult_kernel<<<1, N_DOP, 0, stream>>>(w, old_w, ind, bc, mult);
    pack_w_kernel<<<(UNITS * N_IN) / 256, 256, 0, stream>>>(w, mult, whi, wlo);
    dop_gemm_kernel<<<BATCH / MROWS, 256, 0, stream>>>(x, whi, wlo, b, out);
}
